// SFTLanguageModel_5909875000158
// MI455X (gfx1250) — matrix operand order repaired and checked by reading, not compiled, not run on hardware
//
#include <hip/hip_runtime.h>
#include <hip/hip_bf16.h>

typedef __attribute__((ext_vector_type(16))) _Float16 v16h;
typedef __attribute__((ext_vector_type(8)))  float    v8f;

#define LAYERS  12
#define HDIM    768
#define NHEADS  12
#define FFDIM   3072
#define VOCAB   21128
#define SEQ     512
#define NBATCH  4
#define HEADDIM 64
#define MROWS   (NBATCH*SEQ)   /* 2048 */
#define LN_EPS  1e-12f

// ---------------------------------------------------------------------------
// Block reductions (256 threads)
// ---------------------------------------------------------------------------
__device__ __forceinline__ float block_reduce_sum(float v, float* sh) {
  int t = threadIdx.x;
  sh[t] = v; __syncthreads();
  for (int s = 128; s > 0; s >>= 1) { if (t < s) sh[t] += sh[t + s]; __syncthreads(); }
  float r = sh[0]; __syncthreads();
  return r;
}
__device__ __forceinline__ float block_reduce_max(float v, float* sh) {
  int t = threadIdx.x;
  sh[t] = v; __syncthreads();
  for (int s = 128; s > 0; s >>= 1) { if (t < s) sh[t] = fmaxf(sh[t], sh[t + s]); __syncthreads(); }
  float r = sh[0]; __syncthreads();
  return r;
}

// ---------------------------------------------------------------------------
// Generic batched WMMA GEMM:  C[z] = act( A[z](f16) x op(W[z])(f32->f16) + bias )
//   block tile 128x128, 8 waves, wave tile 64x32 (4x2 16x16 WMMA tiles),
//   K-step 32, double-buffered LDS, software-pipelined K loop.
//   TRANSB=0: op(W)(k,n) = W[k*ldw + n]   TRANSB=1: op(W)(k,n) = W[n*ldw + k]
//   B tile stored K-contiguous per column (lBT[n][k]) so both A and B
//   fragments load as ds_load_b128 pairs.
//   ACT: 0 = none, 1 = exact GELU.
// Requirements (hold for every call): K%32==0, M%128==0, TRANSB => N%128==0.
// ---------------------------------------------------------------------------
template<int ACT, int TRANSB>
__global__ __launch_bounds__(256)
void gemm_wmma(const _Float16* __restrict__ A, const float* __restrict__ W,
               const float* __restrict__ bias,
               float* __restrict__ Cf, _Float16* __restrict__ Ch,
               int M, int N, int K, int lda, int ldw, int ldc, int nh,
               long long sAb, long long sAh, long long sWb, long long sWh,
               long long sCb, long long sCh)
{
  __shared__ _Float16 lA [2][128][40];   // [buf][row m][k]
  __shared__ _Float16 lBT[2][128][40];   // [buf][col n][k]  (transposed B tile)

  const int z  = blockIdx.z;
  const int zb = z / nh, zh = z - zb * nh;
  A += (long long)zb * sAb + (long long)zh * sAh;
  W += (long long)zb * sWb + (long long)zh * sWh;
  const long long coff = (long long)zb * sCb + (long long)zh * sCh;

  const int tid  = threadIdx.x;
  const int lane = tid & 31;
  const int wave = tid >> 5;
  const int wi   = wave >> 2;     // 0..1 : 64-row group
  const int wj   = wave & 3;      // 0..3 : 32-col group
  const int r16  = lane & 15;
  const int hi   = lane >> 4;

  const int blockRow = blockIdx.y * 128;
  const int blockCol = blockIdx.x * 128;

  // staging assignment
  const int ar_st = tid >> 1;           // A row 0..127
  const int ac_st = (tid & 1) << 4;     // A col-chunk 0/16
  const int bn_st = tid & 127;          // B column (n) 0..127
  const int bk_st = (tid >> 7) << 4;    // B k-chunk 0/16
  const int gn    = blockCol + bn_st;   // global n for this thread's B column

  const _Float16* Arow = A + (size_t)(blockRow + ar_st) * lda + ac_st;

  v8f acc[4][2];
  #pragma unroll
  for (int i = 0; i < 4; i++)
    #pragma unroll
    for (int j = 0; j < 2; j++)
      #pragma unroll
      for (int g = 0; g < 8; g++) acc[i][j][g] = 0.f;

  uint4 aS0, aS1;      // staged A (16 f16)
  float bS[16];        // staged B column chunk (16 f32)

  auto loadStage = [&](int k0) {
    const _Float16* as = Arow + k0;
    aS0 = *(const uint4*)as;
    aS1 = *(const uint4*)(as + 8);
    if (TRANSB) {
      // contiguous along k
      const float* wsrc = W + (size_t)gn * ldw + k0 + bk_st;
      #pragma unroll
      for (int j = 0; j < 16; j += 4) {
        float4 f = *(const float4*)(wsrc + j);
        bS[j] = f.x; bS[j+1] = f.y; bS[j+2] = f.z; bS[j+3] = f.w;
      }
    } else {
      if (gn < N) {
        const float* wsrc = W + (size_t)(k0 + bk_st) * ldw + gn;
        #pragma unroll
        for (int j = 0; j < 16; j++) bS[j] = wsrc[(size_t)j * ldw];
        if (k0 + 32 < K) __builtin_prefetch(wsrc + (size_t)32 * ldw, 0, 3);
      } else {
        #pragma unroll
        for (int j = 0; j < 16; j++) bS[j] = 0.f;
      }
    }
  };

  auto storeStage = [&](int buf) {
    *(uint4*)&lA[buf][ar_st][ac_st]     = aS0;
    *(uint4*)&lA[buf][ar_st][ac_st + 8] = aS1;
    #pragma unroll
    for (int j = 0; j < 16; j++)
      lBT[buf][bn_st][bk_st + j] = (_Float16)bS[j];
  };

  auto compute = [&](int buf) {
    // B fragments: 16 K-contiguous halves per lane -> 2x ds_load_b128
    v16h bf[2];
    #pragma unroll
    for (int nt = 0; nt < 2; nt++) {
      int bc = wj * 32 + nt * 16 + r16;
      const _Float16* bp = &lBT[buf][bc][hi << 4];
      #pragma unroll
      for (int j = 0; j < 16; j++) bf[nt][j] = bp[j];
    }
    // A fragments: K pairs 2i+16*hi -> 2x ds_load_b128
    #pragma unroll
    for (int mt = 0; mt < 4; mt++) {
      int arr = wi * 64 + mt * 16 + r16;
      const _Float16* ap = &lA[buf][arr][0];
      v16h af;
      #pragma unroll
      for (int i = 0; i < 8; i++) {
        int k = 2 * i + 16 * hi;
        af[2 * i]     = ap[k];
        af[2 * i + 1] = ap[k + 1];
      }
      #pragma unroll
      for (int nt = 0; nt < 2; nt++) {
        acc[mt][nt] = __builtin_amdgcn_wmma_f32_16x16x32_f16(
            false, af, false, bf[nt], (short)0, acc[mt][nt], false, false);
      }
    }
  };

  // software-pipelined K loop
  loadStage(0);
  storeStage(0);
  __syncthreads();
  const int nk = K >> 5;
  for (int it = 0; it < nk; it++) {
    const int  cur  = it & 1;
    const bool more = (it + 1 < nk);
    if (more) loadStage((it + 1) << 5);   // global loads for next tile
    compute(cur);                         // WMMA on current tile
    if (more) storeStage(cur ^ 1);        // convert + LDS store next tile
    __syncthreads();
  }

  // ---- epilogue: bias, activation, store (C element (m,n) = (g+8*hi, lane&15)) ----
  #pragma unroll
  for (int mt = 0; mt < 4; mt++) {
    #pragma unroll
    for (int nt = 0; nt < 2; nt++) {
      int n = blockCol + wj * 32 + nt * 16 + r16;
      if (n >= N) continue;
      float bval = bias ? bias[n] : 0.f;
      #pragma unroll
      for (int g = 0; g < 8; g++) {
        int m = blockRow + wi * 64 + mt * 16 + g + 8 * hi;
        float v = acc[mt][nt][g] + bval;
        if (ACT == 1) v = 0.5f * v * (1.f + erff(v * 0.70710678118f));
        long long idx = coff + (long long)m * ldc + n;
        if (Cf) Cf[idx] = v;
        if (Ch) Ch[idx] = (_Float16)v;
      }
    }
  }
}

// ---------------------------------------------------------------------------
// Embedding gather + LayerNorm  (one block per token row, 256 thr x 3 cols)
// ---------------------------------------------------------------------------
__global__ __launch_bounds__(256)
void embed_ln_kernel(const int* __restrict__ x, const float* __restrict__ wte,
                     const float* __restrict__ wpe, const float* __restrict__ tte,
                     const float* __restrict__ g, const float* __restrict__ b,
                     float* __restrict__ hF, _Float16* __restrict__ hH)
{
  __shared__ float sh[256];
  int m   = blockIdx.x;
  int s   = m & (SEQ - 1);
  int tid = threadIdx.x;
  int tok = x[m];
  float v[3];
  #pragma unroll
  for (int i = 0; i < 3; i++) {
    int c = tid + i * 256;
    v[i] = wte[(size_t)tok * HDIM + c] + wpe[(size_t)s * HDIM + c] + tte[c];
  }
  float mean = block_reduce_sum(v[0] + v[1] + v[2], sh) * (1.f / HDIM);
  float sq = 0.f;
  #pragma unroll
  for (int i = 0; i < 3; i++) { float d = v[i] - mean; sq += d * d; }
  float var = block_reduce_sum(sq, sh) * (1.f / HDIM);
  float rs = rsqrtf(var + LN_EPS);
  #pragma unroll
  for (int i = 0; i < 3; i++) {
    int c = tid + i * 256;
    float y = (v[i] - mean) * rs * g[c] + b[c];
    hF[(size_t)m * HDIM + c] = y;
    hH[(size_t)m * HDIM + c] = (_Float16)y;
  }
}

// ---------------------------------------------------------------------------
// Residual add + LayerNorm:  h = LN(h + t) ; emits fp32 + f16
// ---------------------------------------------------------------------------
__global__ __launch_bounds__(256)
void add_ln_kernel(float* __restrict__ h, const float* __restrict__ t,
                   const float* __restrict__ g, const float* __restrict__ b,
                   _Float16* __restrict__ hH)
{
  __shared__ float sh[256];
  int m   = blockIdx.x;
  int tid = threadIdx.x;
  float v[3];
  #pragma unroll
  for (int i = 0; i < 3; i++) {
    int c = tid + i * 256;
    v[i] = h[(size_t)m * HDIM + c] + t[(size_t)m * HDIM + c];
  }
  float mean = block_reduce_sum(v[0] + v[1] + v[2], sh) * (1.f / HDIM);
  float sq = 0.f;
  #pragma unroll
  for (int i = 0; i < 3; i++) { float d = v[i] - mean; sq += d * d; }
  float var = block_reduce_sum(sq, sh) * (1.f / HDIM);
  float rs = rsqrtf(var + LN_EPS);
  #pragma unroll
  for (int i = 0; i < 3; i++) {
    int c = tid + i * 256;
    float y = (v[i] - mean) * rs * g[c] + b[c];
    h[(size_t)m * HDIM + c] = y;
    hH[(size_t)m * HDIM + c] = (_Float16)y;
  }
}

// ---------------------------------------------------------------------------
// Prefix-LM masked softmax over S. scores fp32 -> attn f16.
//   grid = (SEQ queries, NBATCH*NHEADS), block = 256 (2 keys per thread)
//   allowed(q,k) = (k < s1) || (q >= s1 && k <= q)
// ---------------------------------------------------------------------------
__global__ __launch_bounds__(256)
void softmax_mask_kernel(const float* __restrict__ scores,
                         _Float16* __restrict__ attn,
                         const int* __restrict__ s1)
{
  __shared__ float sh[256];
  int q   = blockIdx.x;
  int z   = blockIdx.y;           // b*NHEADS + h
  int b   = z / NHEADS;
  int tid = threadIdx.x;
  int sl  = s1[b];
  const float* row = scores + ((size_t)z * SEQ + q) * SEQ;
  float vals[2];
  #pragma unroll
  for (int i = 0; i < 2; i++) {
    int k = tid + i * 256;
    bool allowed = (k < sl) || ((q >= sl) && (k <= q));
    vals[i] = row[k] * 0.125f + (allowed ? 0.f : -1e9f);   // 1/sqrt(64)
  }
  float mx = block_reduce_max(fmaxf(vals[0], vals[1]), sh);
  float e0 = __expf(vals[0] - mx), e1 = __expf(vals[1] - mx);
  float sum = block_reduce_sum(e0 + e1, sh);
  float inv = 1.f / sum;
  _Float16* orow = attn + ((size_t)z * SEQ + q) * SEQ;
  orow[tid]       = (_Float16)(e0 * inv);
  orow[tid + 256] = (_Float16)(e1 * inv);
}

// ---------------------------------------------------------------------------
// In-place softmax over the vocab for each of the 2048 rows of d_out.
// ---------------------------------------------------------------------------
__global__ __launch_bounds__(256)
void vocab_softmax_kernel(float* __restrict__ out)
{
  __shared__ float sh[256];
  int m   = blockIdx.x;
  int tid = threadIdx.x;
  float* row = out + (size_t)m * VOCAB;
  float mx = -3.4e38f;
  for (int k = tid; k < VOCAB; k += 256) mx = fmaxf(mx, row[k]);
  mx = block_reduce_max(mx, sh);
  float s = 0.f;
  for (int k = tid; k < VOCAB; k += 256) s += __expf(row[k] - mx);
  s = block_reduce_sum(s, sh);
  float inv = 1.f / s;
  for (int k = tid; k < VOCAB; k += 256) row[k] = __expf(row[k] - mx) * inv;
}

// ---------------------------------------------------------------------------
// Host orchestration
// ---------------------------------------------------------------------------
extern "C" void kernel_launch(void* const* d_in, const int* in_sizes, int n_in,
                              void* d_out, int out_size, void* d_ws, size_t ws_size,
                              hipStream_t stream)
{
  (void)in_sizes; (void)n_in; (void)out_size; (void)ws_size;
  const int*   x    = (const int*)  d_in[0];
  const int*   s1   = (const int*)  d_in[1];
  const float* wte  = (const float*)d_in[2];
  const float* wpe  = (const float*)d_in[3];
  const float* tte  = (const float*)d_in[4];
  const float* elg  = (const float*)d_in[5];
  const float* elb  = (const float*)d_in[6];
  const float* Wq   = (const float*)d_in[7];
  const float* bq   = (const float*)d_in[8];
  const float* Wk   = (const float*)d_in[9];
  const float* bk   = (const float*)d_in[10];
  const float* Wv   = (const float*)d_in[11];
  const float* bv   = (const float*)d_in[12];
  const float* Wo   = (const float*)d_in[13];
  const float* bo   = (const float*)d_in[14];
  const float* ln1g = (const float*)d_in[15];
  const float* ln1b = (const float*)d_in[16];
  const float* Wf1  = (const float*)d_in[17];
  const float* bf1  = (const float*)d_in[18];
  const float* Wf2  = (const float*)d_in[19];
  const float* bf2  = (const float*)d_in[20];
  const float* ln2g = (const float*)d_in[21];
  const float* ln2b = (const float*)d_in[22];
  const float* Wc   = (const float*)d_in[23];
  const float* bc   = (const float*)d_in[24];
  float* out = (float*)d_out;

  char* ws = (char*)d_ws;
  size_t off = 0;
  auto alloc = [&](size_t bytes) -> char* {
    char* p = ws + off;
    off = (off + bytes + 255) & ~(size_t)255;
    return p;
  };
  float*    hF   = (float*)   alloc((size_t)MROWS * HDIM * 4);
  _Float16* hH   = (_Float16*)alloc((size_t)MROWS * HDIM * 2);
  _Float16* qH   = (_Float16*)alloc((size_t)MROWS * HDIM * 2);
  float*    kF   = (float*)   alloc((size_t)MROWS * HDIM * 4);
  float*    vF   = (float*)   alloc((size_t)MROWS * HDIM * 4);
  float*    tmpF = (float*)   alloc((size_t)MROWS * HDIM * 4);
  float*    scF  = (float*)   alloc((size_t)NBATCH * NHEADS * SEQ * SEQ * 4);
  _Float16* atH  = (_Float16*)alloc((size_t)NBATCH * NHEADS * SEQ * SEQ * 2);
  _Float16* ctxH = (_Float16*)alloc((size_t)MROWS * HDIM * 2);
  _Float16* ffH  = (_Float16*)alloc((size_t)MROWS * FFDIM * 2);

  embed_ln_kernel<<<MROWS, 256, 0, stream>>>(x, wte, wpe, tte, elg, elb, hF, hH);

  const dim3 gHH(HDIM / 128, MROWS / 128, 1);               // M=2048, N=768
  const dim3 gFF1(FFDIM / 128, MROWS / 128, 1);             // N=3072
  const dim3 gSc(SEQ / 128, SEQ / 128, NBATCH * NHEADS);    // 4x4x48
  const dim3 gCtx(1, SEQ / 128, NBATCH * NHEADS);           // N=64
  const dim3 gCls((VOCAB + 127) / 128, MROWS / 128, 1);     // N=21128

  for (int l = 0; l < LAYERS; l++) {
    const float* wq = Wq + (size_t)l * HDIM * HDIM;
    const float* wk = Wk + (size_t)l * HDIM * HDIM;
    const float* wv = Wv + (size_t)l * HDIM * HDIM;
    const float* wo = Wo + (size_t)l * HDIM * HDIM;
    const float* w1 = Wf1 + (size_t)l * HDIM * FFDIM;
    const float* w2 = Wf2 + (size_t)l * FFDIM * HDIM;

    // q/k/v projections (q -> f16 only, k/v -> fp32 for on-the-fly B conversion)
    gemm_wmma<0,0><<<gHH, 256, 0, stream>>>(hH, wq, bq + l * HDIM, nullptr, qH,
        MROWS, HDIM, HDIM, HDIM, HDIM, HDIM, 1, 0,0, 0,0, 0,0);
    gemm_wmma<0,0><<<gHH, 256, 0, stream>>>(hH, wk, bk + l * HDIM, kF, nullptr,
        MROWS, HDIM, HDIM, HDIM, HDIM, HDIM, 1, 0,0, 0,0, 0,0);
    gemm_wmma<0,0><<<gHH, 256, 0, stream>>>(hH, wv, bv + l * HDIM, vF, nullptr,
        MROWS, HDIM, HDIM, HDIM, HDIM, HDIM, 1, 0,0, 0,0, 0,0);

    // scores[z] = Q[b,:,h,:] x K[b,:,h,:]^T   (z = b*NHEADS + h)
    gemm_wmma<0,1><<<gSc, 256, 0, stream>>>(qH, kF, nullptr, scF, nullptr,
        SEQ, SEQ, HEADDIM, HDIM, HDIM, SEQ, NHEADS,
        (long long)SEQ * HDIM, (long long)HEADDIM,
        (long long)SEQ * HDIM, (long long)HEADDIM,
        (long long)NHEADS * SEQ * SEQ, (long long)SEQ * SEQ);

    softmax_mask_kernel<<<dim3(SEQ, NBATCH * NHEADS), 256, 0, stream>>>(scF, atH, s1);

    // ctx[z] = attn[z] x V[b,:,h,:]   -> f16 [2048, 768]
    gemm_wmma<0,0><<<gCtx, 256, 0, stream>>>(atH, vF, nullptr, nullptr, ctxH,
        SEQ, HEADDIM, SEQ, SEQ, HDIM, HDIM, NHEADS,
        (long long)NHEADS * SEQ * SEQ, (long long)SEQ * SEQ,
        (long long)SEQ * HDIM, (long long)HEADDIM,
        (long long)SEQ * HDIM, (long long)HEADDIM);

    // output projection + residual LN
    gemm_wmma<0,0><<<gHH, 256, 0, stream>>>(ctxH, wo, bo + l * HDIM, tmpF, nullptr,
        MROWS, HDIM, HDIM, HDIM, HDIM, HDIM, 1, 0,0, 0,0, 0,0);
    add_ln_kernel<<<MROWS, 256, 0, stream>>>(hF, tmpF, ln1g + l * HDIM, ln1b + l * HDIM, hH);

    // FFN: GELU(h W1 + b1) W2 + b2, residual LN
    gemm_wmma<1,0><<<gFF1, 256, 0, stream>>>(hH, w1, bf1 + l * FFDIM, nullptr, ffH,
        MROWS, FFDIM, HDIM, HDIM, FFDIM, FFDIM, 1, 0,0, 0,0, 0,0);
    gemm_wmma<0,0><<<gHH, 256, 0, stream>>>(ffH, w2, bf2 + l * HDIM, tmpF, nullptr,
        MROWS, HDIM, FFDIM, FFDIM, HDIM, HDIM, 1, 0,0, 0,0, 0,0);
    add_ln_kernel<<<MROWS, 256, 0, stream>>>(hF, tmpF, ln2g + l * HDIM, ln2b + l * HDIM, hH);
  }

  // classifier -> logits directly into d_out, then softmax in place
  gemm_wmma<0,0><<<gCls, 256, 0, stream>>>(hH, Wc, bc, out, nullptr,
      MROWS, VOCAB, HDIM, HDIM, VOCAB, VOCAB, 1, 0,0, 0,0, 0,0);
  vocab_softmax_kernel<<<MROWS, 256, 0, stream>>>(out);
}
